// ResidualGNN_58634893525193
// MI455X (gfx1250) — matrix operand order repaired and checked by reading, not compiled, not run on hardware
//
#include <hip/hip_runtime.h>

#define N_NODES 50000
#define N_EDGES 500000
#define IN_DIM 64
#define EDGE_DIM 32
#define HID 128
#define N_LAYERS 4

typedef __attribute__((ext_vector_type(16))) _Float16 v16h;
typedef __attribute__((ext_vector_type(8)))  _Float16 v8h;
typedef __attribute__((ext_vector_type(8)))  float    v8f;

// LDS halves: w1 tile + w2 tile + 8 per-wave m1 tiles
#define EDGE_LDS_HALVES (HID * (HID + EDGE_DIM) + HID * HID + 8 * 16 * HID)
#define EDGE_LDS_BYTES  (EDGE_LDS_HALVES * 2)
#define GRU_LDS_HALVES  (2 * 3 * HID * HID)
#define GRU_LDS_BYTES   (GRU_LDS_HALVES * 2)

__device__ __forceinline__ v8f wmma16(v16h a, v16h b, v8f c) {
  // D = A(16x32 f16) * B(32x16 f16) + C(16x16 f32)
  return __builtin_amdgcn_wmma_f32_16x16x32_f16(false, a, false, b, (short)0, c,
                                                false, false);
}

__device__ __forceinline__ v8f zero8() {
  v8f r;
#pragma unroll
  for (int i = 0; i < 8; ++i) r[i] = 0.0f;
  return r;
}

// fast activations: v_exp_f32 + v_rcp_f32, no IEEE division expansion
__device__ __forceinline__ float sigmoid_f(float v) {
  return __builtin_amdgcn_rcpf(1.0f + __expf(-v));
}
__device__ __forceinline__ float silu_f(float v) {
  return v * __builtin_amdgcn_rcpf(1.0f + __expf(-v));
}
__device__ __forceinline__ float tanh_f(float v) {
  return 2.0f * sigmoid_f(2.0f * v) - 1.0f;
}

__device__ __forceinline__ void atom_add_f32(float* p, float v) {
  unsafeAtomicAdd(p, v);
}

// A fragment (16x32, f16) built from an fp32 row in memory.
// CDNA5 ISA 7.12.2: lane<16 holds row=lane, K = {kb..kb+7, kb+16..kb+23}, kb=0;
// lane>=16 same rows with kb=8.  kBase = 32*chunk + 8*(lane>>4).
__device__ __forceinline__ v16h a_frag_f32row(const float* __restrict__ row, int kBase) {
  v16h r;
#pragma unroll
  for (int i = 0; i < 8; ++i) {
    r[i]     = (_Float16)row[kBase + i];
    r[i + 8] = (_Float16)row[kBase + 16 + i];
  }
  return r;
}

// A fragment directly from an f16 row (two aligned 16B loads, no converts).
__device__ __forceinline__ v16h a_frag_f16row(const _Float16* __restrict__ row, int kBase) {
  v8h lo = *(const v8h*)(row + kBase);
  v8h hi = *(const v8h*)(row + kBase + 16);
  v16h r;
#pragma unroll
  for (int i = 0; i < 8; ++i) { r[i] = lo[i]; r[i + 8] = hi[i]; }
  return r;
}

// B fragment (32x16, f16) of W^T where W is [Nout x K] row-major f16
// (works for global or LDS-resident W; address space inferred per call site).
// Lane holds column n = colBase + (lane&15); K = {kb..kb+7, kb+16..kb+23}, kb = 8*(lane>>4).
__device__ __forceinline__ v16h b_frag(const _Float16* __restrict__ w, int K,
                                       int colBase, int kChunk, int lane) {
  const _Float16* p =
      w + (size_t)(colBase + (lane & 15)) * K + kChunk + ((lane & 16) ? 8 : 0);
  v8h lo = *(const v8h*)p;
  v8h hi = *(const v8h*)(p + 16);
  v16h r;
#pragma unroll
  for (int i = 0; i < 8; ++i) { r[i] = lo[i]; r[i + 8] = hi[i]; }
  return r;
}

// ---------------------------------------------------------------- utilities
__global__ void cvt_f32_to_f16_kernel(const float* __restrict__ s,
                                      _Float16* __restrict__ d, int n) {
  int i = blockIdx.x * blockDim.x + threadIdx.x;
  if (i < n) d[i] = (_Float16)s[i];
}

__global__ void zero_f32_kernel(float* __restrict__ p, int n) {
  int i = blockIdx.x * blockDim.x + threadIdx.x;
  if (i < n) p[i] = 0.0f;
}

__global__ void init_out_kernel(float* __restrict__ out,
                                const float* __restrict__ b2) {
  out[0] = (float)N_NODES * b2[0];
}

// ------------------------------------------------------------------ embed
// h = silu(x @ embed_w^T + embed_b); one 16-node tile per wave.
__global__ __launch_bounds__(256) void embed_kernel(
    const float* __restrict__ x, const _Float16* __restrict__ w,  // [HID x IN_DIM] f16
    const float* __restrict__ b, float* __restrict__ h) {
  const int lane = threadIdx.x & 31;
  const int half = lane >> 4, l16 = lane & 15;
  const int waveId = (blockIdx.x * blockDim.x + threadIdx.x) >> 5;
  const int wavesTotal = (gridDim.x * blockDim.x) >> 5;
  const int nTiles = N_NODES / 16;

  for (int tile = waveId; tile < nTiles; tile += wavesTotal) {
    const float* xrow = x + (size_t)(tile * 16 + l16) * IN_DIM;
    v16h a[2];
#pragma unroll
    for (int c = 0; c < 2; ++c) a[c] = a_frag_f32row(xrow, c * 32 + half * 8);
    for (int ot = 0; ot < 8; ++ot) {
      v8f acc = zero8();
#pragma unroll
      for (int c = 0; c < 2; ++c)
        acc = wmma16(a[c], b_frag(w, IN_DIM, ot * 16, c * 32, lane), acc);
      const int n = ot * 16 + l16;
      const float bias = b[n];
#pragma unroll
      for (int j = 0; j < 8; ++j) {
        float v = silu_f(acc[j] + bias);
        h[(size_t)(tile * 16 + j + half * 8) * HID + n] = v;
      }
    }
  }
}

// ------------------------------------------------------------- edge message
// m = silu(silu([h[src], eattr] @ W1^T + b1) @ W2^T + b2); scatter-add to agg[dst].
// Weights staged in LDS once per workgroup; B-fragments served by ds_load_b128.
__global__ __launch_bounds__(256) void edge_msg_kernel(
    const _Float16* __restrict__ h16, const int* __restrict__ edge_index,
    const float* __restrict__ edge_attr,
    const _Float16* __restrict__ w1g, const float* __restrict__ b1,  // [HID x 160]
    const _Float16* __restrict__ w2g, const float* __restrict__ b2,  // [HID x HID]
    float* __restrict__ agg) {
  extern __shared__ _Float16 esmem[];
  _Float16* sw1 = esmem;                          // 128*160 halves (40 KB)
  _Float16* sw2 = sw1 + HID * (HID + EDGE_DIM);   // 128*128 halves (32 KB)
  _Float16* sm1 = sw2 + HID * HID;                // 8 waves * 16*128 halves (32 KB)

  // cooperative weight staging (b128 granules)
  {
    v8h* d1 = (v8h*)sw1; const v8h* s1 = (const v8h*)w1g;
    for (int i = threadIdx.x; i < HID * (HID + EDGE_DIM) / 8; i += blockDim.x) d1[i] = s1[i];
    v8h* d2 = (v8h*)sw2; const v8h* s2 = (const v8h*)w2g;
    for (int i = threadIdx.x; i < HID * HID / 8; i += blockDim.x) d2[i] = s2[i];
  }
  __syncthreads();

  const int lane = threadIdx.x & 31;
  const int wave = threadIdx.x >> 5;
  const int half = lane >> 4, l16 = lane & 15;
  const int waveId = (blockIdx.x * blockDim.x + threadIdx.x) >> 5;
  const int wavesTotal = (gridDim.x * blockDim.x) >> 5;
  const int nTiles = N_EDGES / 16;
  const int KIN = HID + EDGE_DIM;  // 160
  _Float16* mtile = sm1 + wave * (16 * HID);

  for (int tile = waveId; tile < nTiles; tile += wavesTotal) {
    const int e0 = tile * 16;
    const int srcIdx = edge_index[e0 + l16];  // source node of row l16
    const _Float16* hrow = h16 + (size_t)srcIdx * HID;
    const float* erow = edge_attr + (size_t)(e0 + l16) * EDGE_DIM;

    // ---- GEMM1: K = 160 = 4 chunks of h + 1 chunk of edge_attr ----
    v8f acc[8] = {};
#pragma unroll
    for (int c = 0; c < 5; ++c) {
      v16h a = (c < 4) ? a_frag_f16row(hrow, c * 32 + half * 8)
                       : a_frag_f32row(erow, half * 8);
      for (int ot = 0; ot < 8; ++ot)
        acc[ot] = wmma16(a, b_frag(sw1, KIN, ot * 16, c * 32, lane), acc[ot]);
    }
    // bias + silu, stash f16 tile (row-major) in LDS for the transpose
    for (int ot = 0; ot < 8; ++ot) {
      const int n = ot * 16 + l16;
      const float bias = b1[n];
#pragma unroll
      for (int j = 0; j < 8; ++j) {
        float v = silu_f(acc[ot][j] + bias);
        mtile[(j + half * 8) * HID + n] = (_Float16)v;
      }
    }
    asm volatile("s_wait_dscnt 0" ::: "memory");

    // ---- GEMM2: re-load tile in A layout, K = 128 ----
    const _Float16* mrow = mtile + l16 * HID;
    v16h a2[4];
#pragma unroll
    for (int c = 0; c < 4; ++c) a2[c] = a_frag_f16row(mrow, c * 32 + half * 8);

    v8f acc2[8] = {};
    for (int ot = 0; ot < 8; ++ot)
#pragma unroll
      for (int c = 0; c < 4; ++c)
        acc2[ot] = wmma16(a2[c], b_frag(sw2, HID, ot * 16, c * 32, lane), acc2[ot]);

    // scatter-add: row m = j + 8*half, col n = 16*ot + l16
    int dstIdx[8];
#pragma unroll
    for (int j = 0; j < 8; ++j)
      dstIdx[j] = edge_index[N_EDGES + e0 + j + half * 8];
    for (int ot = 0; ot < 8; ++ot) {
      const int n = ot * 16 + l16;
      const float bias = b2[n];
#pragma unroll
      for (int j = 0; j < 8; ++j) {
        float v = silu_f(acc2[ot][j] + bias);
        atom_add_f32(&agg[(size_t)dstIdx[j] * HID + n], v);
      }
    }
  }
}

// ---------------------------------------------------------------- GRU cell
// gi = agg @ Wih^T + bih ; gh = h @ Whh^T + bhh ; standard GRU update, in place.
// Wih/Whh (192 KB f16) staged in LDS once per workgroup.
__global__ __launch_bounds__(256) void gru_kernel(
    const float* __restrict__ agg, float* __restrict__ h,
    const _Float16* __restrict__ h16,
    const _Float16* __restrict__ wihg, const _Float16* __restrict__ whhg,  // [384 x 128]
    const float* __restrict__ bih, const float* __restrict__ bhh) {
  extern __shared__ _Float16 gsmem[];
  _Float16* swih = gsmem;                 // 384*128 halves (96 KB)
  _Float16* swhh = gsmem + 3 * HID * HID; // 384*128 halves (96 KB)
  {
    v8h* d1 = (v8h*)swih; const v8h* s1 = (const v8h*)wihg;
    for (int i = threadIdx.x; i < 3 * HID * HID / 8; i += blockDim.x) d1[i] = s1[i];
    v8h* d2 = (v8h*)swhh; const v8h* s2 = (const v8h*)whhg;
    for (int i = threadIdx.x; i < 3 * HID * HID / 8; i += blockDim.x) d2[i] = s2[i];
  }
  __syncthreads();

  const int lane = threadIdx.x & 31;
  const int half = lane >> 4, l16 = lane & 15;
  const int waveId = (blockIdx.x * blockDim.x + threadIdx.x) >> 5;
  const int wavesTotal = (gridDim.x * blockDim.x) >> 5;
  const int nTiles = N_NODES / 16;

  for (int tile = waveId; tile < nTiles; tile += wavesTotal) {
    const float* arow = agg + (size_t)(tile * 16 + l16) * HID;
    const _Float16* hrow = h16 + (size_t)(tile * 16 + l16) * HID;
    v16h aA[4], hA[4];
#pragma unroll
    for (int c = 0; c < 4; ++c) {
      aA[c] = a_frag_f32row(arow, c * 32 + half * 8);
      hA[c] = a_frag_f16row(hrow, c * 32 + half * 8);
    }
    for (int ot = 0; ot < 8; ++ot) {
      v8f gir = zero8(), ghr = zero8(), giz = zero8(), ghz = zero8(),
          gin = zero8(), ghn = zero8();
#pragma unroll
      for (int c = 0; c < 4; ++c) {
        gir = wmma16(aA[c], b_frag(swih, HID, ot * 16, c * 32, lane), gir);
        ghr = wmma16(hA[c], b_frag(swhh, HID, ot * 16, c * 32, lane), ghr);
        giz = wmma16(aA[c], b_frag(swih, HID, 128 + ot * 16, c * 32, lane), giz);
        ghz = wmma16(hA[c], b_frag(swhh, HID, 128 + ot * 16, c * 32, lane), ghz);
        gin = wmma16(aA[c], b_frag(swih, HID, 256 + ot * 16, c * 32, lane), gin);
        ghn = wmma16(hA[c], b_frag(swhh, HID, 256 + ot * 16, c * 32, lane), ghn);
      }
      const int n = ot * 16 + l16;
      const float bir = bih[n], bhr = bhh[n];
      const float biz = bih[128 + n], bhz = bhh[128 + n];
      const float bin_ = bih[256 + n], bhn = bhh[256 + n];
#pragma unroll
      for (int j = 0; j < 8; ++j) {
        float r = sigmoid_f(gir[j] + bir + ghr[j] + bhr);
        float z = sigmoid_f(giz[j] + biz + ghz[j] + bhz);
        float nn = tanh_f(gin[j] + bin_ + r * (ghn[j] + bhn));
        const size_t idx = (size_t)(tile * 16 + j + half * 8) * HID + n;
        float hold = h[idx];
        h[idx] = (1.0f - z) * nn + z * hold;
      }
    }
  }
}

// ------------------------------------------------------------------- head
// out = sum_m [ silu(h @ W1^T + b1) @ w2 + b2 ]  (b2*N added by init_out_kernel)
__global__ __launch_bounds__(256) void head_kernel(
    const _Float16* __restrict__ h16, const _Float16* __restrict__ w1,  // [128 x 128]
    const float* __restrict__ b1, const float* __restrict__ w2,
    float* __restrict__ out) {
  const int lane = threadIdx.x & 31;
  const int half = lane >> 4, l16 = lane & 15;
  const int waveId = (blockIdx.x * blockDim.x + threadIdx.x) >> 5;
  const int wavesTotal = (gridDim.x * blockDim.x) >> 5;
  const int nTiles = N_NODES / 16;

  float psum = 0.0f;
  for (int tile = waveId; tile < nTiles; tile += wavesTotal) {
    const _Float16* hrow = h16 + (size_t)(tile * 16 + l16) * HID;
    v16h hA[4];
#pragma unroll
    for (int c = 0; c < 4; ++c) hA[c] = a_frag_f16row(hrow, c * 32 + half * 8);
    for (int ot = 0; ot < 8; ++ot) {
      v8f acc = zero8();
#pragma unroll
      for (int c = 0; c < 4; ++c)
        acc = wmma16(hA[c], b_frag(w1, HID, ot * 16, c * 32, lane), acc);
      const int n = ot * 16 + l16;
      const float bias = b1[n], wn = w2[n];
#pragma unroll
      for (int j = 0; j < 8; ++j) psum += silu_f(acc[j] + bias) * wn;
    }
  }
  // wave32 reduction; every (lane, vgpr, ot) element of the tiles is unique.
#pragma unroll
  for (int off = 16; off > 0; off >>= 1) psum += __shfl_xor(psum, off, 32);
  if (lane == 0) atom_add_f32(out, psum);
}

// ------------------------------------------------------------------ launch
static inline size_t align256(size_t x) { return (x + 255) & ~(size_t)255; }

extern "C" void kernel_launch(void* const* d_in, const int* in_sizes, int n_in,
                              void* d_out, int out_size, void* d_ws, size_t ws_size,
                              hipStream_t stream) {
  const float* x        = (const float*)d_in[0];
  const int*   ei       = (const int*)d_in[1];
  const float* ea       = (const float*)d_in[2];
  const float* embed_w  = (const float*)d_in[3];
  const float* embed_b  = (const float*)d_in[4];
  const float* msg_w1   = (const float*)d_in[5];
  const float* msg_b1   = (const float*)d_in[6];
  const float* msg_w2   = (const float*)d_in[7];
  const float* msg_b2   = (const float*)d_in[8];
  const float* gru_wih  = (const float*)d_in[9];
  const float* gru_whh  = (const float*)d_in[10];
  const float* gru_bih  = (const float*)d_in[11];
  const float* gru_bhh  = (const float*)d_in[12];
  const float* head_w1  = (const float*)d_in[13];
  const float* head_b1  = (const float*)d_in[14];
  const float* head_w2  = (const float*)d_in[15];
  const float* head_b2  = (const float*)d_in[16];

  const int nEmbedW = HID * IN_DIM;
  const int nMsgW1  = N_LAYERS * HID * (HID + EDGE_DIM);
  const int nMsgW2  = N_LAYERS * HID * HID;
  const int nGruW   = N_LAYERS * 3 * HID * HID;
  const int nHeadW1 = HID * HID;
  const int nH      = N_NODES * HID;

  char* ws = (char*)d_ws;
  float*    h    = (float*)ws;    ws += align256((size_t)nH * sizeof(float));
  float*    agg  = (float*)ws;    ws += align256((size_t)nH * sizeof(float));
  _Float16* h16  = (_Float16*)ws; ws += align256((size_t)nH * 2);
  _Float16* f_embed = (_Float16*)ws; ws += align256((size_t)nEmbedW * 2);
  _Float16* f_m1    = (_Float16*)ws; ws += align256((size_t)nMsgW1 * 2);
  _Float16* f_m2    = (_Float16*)ws; ws += align256((size_t)nMsgW2 * 2);
  _Float16* f_gih   = (_Float16*)ws; ws += align256((size_t)nGruW * 2);
  _Float16* f_ghh   = (_Float16*)ws; ws += align256((size_t)nGruW * 2);
  _Float16* f_h1    = (_Float16*)ws; ws += align256((size_t)nHeadW1 * 2);

  // allow large dynamic LDS (WGP has 320 KB); idempotent + deterministic
  hipFuncSetAttribute((const void*)edge_msg_kernel,
                      hipFuncAttributeMaxDynamicSharedMemorySize, EDGE_LDS_BYTES);
  hipFuncSetAttribute((const void*)gru_kernel,
                      hipFuncAttributeMaxDynamicSharedMemorySize, GRU_LDS_BYTES);

  // weight conversion (fp32 -> f16)
  cvt_f32_to_f16_kernel<<<(nEmbedW + 255) / 256, 256, 0, stream>>>(embed_w, f_embed, nEmbedW);
  cvt_f32_to_f16_kernel<<<(nMsgW1 + 255) / 256, 256, 0, stream>>>(msg_w1, f_m1, nMsgW1);
  cvt_f32_to_f16_kernel<<<(nMsgW2 + 255) / 256, 256, 0, stream>>>(msg_w2, f_m2, nMsgW2);
  cvt_f32_to_f16_kernel<<<(nGruW + 255) / 256, 256, 0, stream>>>(gru_wih, f_gih, nGruW);
  cvt_f32_to_f16_kernel<<<(nGruW + 255) / 256, 256, 0, stream>>>(gru_whh, f_ghh, nGruW);
  cvt_f32_to_f16_kernel<<<(nHeadW1 + 255) / 256, 256, 0, stream>>>(head_w1, f_h1, nHeadW1);

  init_out_kernel<<<1, 1, 0, stream>>>((float*)d_out, head_b2);

  embed_kernel<<<256, 256, 0, stream>>>(x, f_embed, embed_b, h);
  cvt_f32_to_f16_kernel<<<(nH + 255) / 256, 256, 0, stream>>>(h, h16, nH);

  for (int l = 0; l < N_LAYERS; ++l) {
    zero_f32_kernel<<<(nH + 255) / 256, 256, 0, stream>>>(agg, nH);
    edge_msg_kernel<<<512, 256, EDGE_LDS_BYTES, stream>>>(
        h16, ei, ea,
        f_m1 + (size_t)l * HID * (HID + EDGE_DIM), msg_b1 + l * HID,
        f_m2 + (size_t)l * HID * HID,              msg_b2 + l * HID, agg);
    gru_kernel<<<256, 256, GRU_LDS_BYTES, stream>>>(
        agg, h, h16,
        f_gih + (size_t)l * 3 * HID * HID, f_ghh + (size_t)l * 3 * HID * HID,
        gru_bih + l * 3 * HID, gru_bhh + l * 3 * HID);
    cvt_f32_to_f16_kernel<<<(nH + 255) / 256, 256, 0, stream>>>(h, h16, nH);
  }

  head_kernel<<<256, 256, 0, stream>>>(h16, f_h1, head_b1, head_w2, (float*)d_out);
}
